// MultiOmicsGenerator_36919538876538
// MI455X (gfx1250) — matrix operand order repaired and checked by reading, not compiled, not run on hardware
//
#include <hip/hip_runtime.h>
#include <stdint.h>

typedef __attribute__((ext_vector_type(16))) __bf16 v16bf;
typedef __attribute__((ext_vector_type(8)))  float  v8f;

__device__ __forceinline__ unsigned short f2bf(float f) {
  unsigned int u = __float_as_uint(f);
  unsigned int r = 0x7FFFu + ((u >> 16) & 1u);   // round-to-nearest-even
  return (unsigned short)((u + r) >> 16);
}

// ---------------- elementwise / reduction kernels ----------------

__global__ void k_sum_nodes(const float* __restrict__ x, unsigned short* __restrict__ o, int n) {
  int i = blockIdx.x * 256 + threadIdx.x;
  if (i < n) {
    float s = 0.25f * (x[i] + x[i + n] + x[i + 2 * n] + x[i + 3 * n]);
    o[i] = f2bf(s);
  }
}

__global__ void k_cvt_bf16(const float* __restrict__ x, unsigned short* __restrict__ o, int n) {
  int i = blockIdx.x * 256 + threadIdx.x;
  if (i < n) o[i] = f2bf(x[i]);
}

__global__ void k_bias_relu_bf16(const float* __restrict__ x, const float* __restrict__ b,
                                 unsigned short* __restrict__ o, int n, int C) {
  int i = blockIdx.x * 256 + threadIdx.x;
  if (i < n) {
    float v = x[i] + b[i % C];
    o[i] = f2bf(v > 0.f ? v : 0.f);
  }
}

__global__ void k_colstats_partial(const float* __restrict__ x, float* __restrict__ part,
                                   int C, int rowsPerChunk, int nchunks) {
  int col = blockIdx.x * 256 + threadIdx.x;
  if (col >= C) return;
  int ch = blockIdx.y;
  int r0 = ch * rowsPerChunk;
  float s = 0.f, s2 = 0.f;
  for (int r = r0; r < r0 + rowsPerChunk; ++r) {
    float v = x[(size_t)r * C + col];
    s += v; s2 += v * v;
  }
  part[(size_t)ch * C + col] = s;
  part[(size_t)(nchunks + ch) * C + col] = s2;
}

__global__ void k_colstats_finalize(const float* __restrict__ part, float* __restrict__ stats,
                                    int C, int nchunks) {
  int col = blockIdx.x * 256 + threadIdx.x;
  if (col >= C) return;
  float s = 0.f, s2 = 0.f;
  for (int ch = 0; ch < nchunks; ++ch) {
    s  += part[(size_t)ch * C + col];
    s2 += part[(size_t)(nchunks + ch) * C + col];
  }
  stats[col] = s;
  stats[C + col] = s2;
}

__global__ void k_bn_relu_bf16(const float* __restrict__ x, const float* __restrict__ stats,
                               const float* __restrict__ g, const float* __restrict__ be,
                               unsigned short* __restrict__ o, int n, int C, float invM) {
  int i = blockIdx.x * 256 + threadIdx.x;
  if (i >= n) return;
  int c = i % C;
  float mu  = stats[c] * invM;
  float var = stats[C + c] * invM - mu * mu;
  float v = (x[i] - mu) * rsqrtf(var + 1e-5f) * g[c] + be[c];
  o[i] = f2bf(v > 0.f ? v : 0.f);
}

__global__ void k_bn_inplace(float* __restrict__ x, const float* __restrict__ stats,
                             const float* __restrict__ g, const float* __restrict__ be,
                             int n, int C, float invM) {
  int i = blockIdx.x * 256 + threadIdx.x;
  if (i >= n) return;
  int c = i % C;
  float mu  = stats[c] * invM;
  float var = stats[C + c] * invM - mu * mu;
  x[i] = (x[i] - mu) * rsqrtf(var + 1e-5f) * g[c] + be[c];
}

// ---------------- WMMA bf16 GEMM: C[M,N] = A[M,K] * B[K,N] ----------------
// 256 threads = 8 waves; block tile 128x128, K-step 32.
// LDS holds K-pair-interleaved u32 so fragments load with plain 32-bit ds reads
// giving A and B fragments the same contiguous k order: elem s of lane half h -> k = 16h + s.

#define BM 128
#define BN 128
#define BK 32

union FragU { unsigned int u[8]; v16bf v; };

__global__ __launch_bounds__(256) void k_gemm_bf16(
    const unsigned short* __restrict__ A,   // [M,K] bf16 row-major
    const unsigned short* __restrict__ B,   // [K,N] bf16 row-major
    float* __restrict__ C,                  // [M,N] f32 row-major
    int M, int N, int K)
{
  __shared__ unsigned int AsP[BM][BK / 2 + 1];   // 128 x 17 u32 (stride 17 banks: conflict-free)
  __shared__ unsigned int BsP[BK / 2][BN + 4];   // 16 x 132 u32

  const int tid  = threadIdx.x;
  const int wave = tid >> 5;
  const int lane = tid & 31;
  const int half = lane >> 4;
  const int l16  = lane & 15;
  const int rowBase = blockIdx.y * BM;
  const int colBase = blockIdx.x * BN;

  v8f acc[8];
#pragma unroll
  for (int i = 0; i < 8; ++i)
#pragma unroll
    for (int j = 0; j < 8; ++j) acc[i][j] = 0.f;

  for (int kt = 0; kt < K; kt += BK) {
    // ---- stage A tile (BM x BK): pairs along K are naturally adjacent ----
#pragma unroll
    for (int it = 0; it < 2; ++it) {
      int q  = tid + it * 256;          // 0..511
      int r  = q >> 2;                  // 0..127
      int c4 = q & 3;                   // group of 8 bf16
      uint4 v = *(const uint4*)(A + (size_t)(rowBase + r) * K + kt + c4 * 8);
      AsP[r][c4 * 4 + 0] = v.x;
      AsP[r][c4 * 4 + 1] = v.y;
      AsP[r][c4 * 4 + 2] = v.z;
      AsP[r][c4 * 4 + 3] = v.w;
    }
    // ---- stage B tile (BK x BN): interleave rows k,k+1 into u32 pairs ----
    {
      int kp = tid >> 4;                // 0..15
      int c8 = (tid & 15) * 8;          // 0..120
      int gc = colBase + c8;
      uint4 v0 = {0, 0, 0, 0}, v1 = {0, 0, 0, 0};
      if (gc < N) {                     // N always multiple of 8 -> whole uint4 in/out
        v0 = *(const uint4*)(B + (size_t)(kt + 2 * kp)     * N + gc);
        v1 = *(const uint4*)(B + (size_t)(kt + 2 * kp + 1) * N + gc);
      }
      unsigned int lo[4] = {v0.x, v0.y, v0.z, v0.w};
      unsigned int hi[4] = {v1.x, v1.y, v1.z, v1.w};
#pragma unroll
      for (int j = 0; j < 4; ++j) {
        BsP[kp][c8 + 2 * j]     = (lo[j] & 0x0000FFFFu) | (hi[j] << 16);
        BsP[kp][c8 + 2 * j + 1] = (lo[j] >> 16)         | (hi[j] & 0xFFFF0000u);
      }
    }
    __syncthreads();

    // A fragment: lane(m=l16, half); elem e -> kpair = e + 8*half
    FragU fa;
#pragma unroll
    for (int e = 0; e < 8; ++e) {
      int kp = e + half * 8;
      fa.u[e] = AsP[wave * 16 + l16][kp];
    }
#pragma unroll
    for (int nt = 0; nt < 8; ++nt) {
      // B fragment: lane(n=l16, half); elem e -> kpair row = e + 8*half
      FragU fb;
#pragma unroll
      for (int e = 0; e < 8; ++e)
        fb.u[e] = BsP[e + half * 8][nt * 16 + l16];
      acc[nt] = __builtin_amdgcn_wmma_f32_16x16x32_bf16(
          false, fa.v, false, fb.v, (short)0, acc[nt], false, false);
    }
    __syncthreads();
  }

  // ---- store C: VGPR e holds M = e + 8*half, N = l16 within each 16x16 tile ----
#pragma unroll
  for (int nt = 0; nt < 8; ++nt) {
    int col = colBase + nt * 16 + l16;
    if (col < N) {
      int row0 = rowBase + wave * 16 + half * 8;
#pragma unroll
      for (int e = 0; e < 8; ++e)
        C[(size_t)(row0 + e) * N + col] = acc[nt][e];
    }
  }
}

// ---------------- host orchestration ----------------

extern "C" void kernel_launch(void* const* d_in, const int* in_sizes, int n_in,
                              void* d_out, int out_size, void* d_ws, size_t ws_size,
                              hipStream_t stream) {
  (void)in_sizes; (void)n_in; (void)out_size; (void)ws_size;
  const int M = 8192, L = 128, H = 512;
  const int NCH = 32;                       // row chunks for column stats
  const int dims[4] = {5000, 1000, 5000, 2000};

  char* ws = (char*)d_ws;
  size_t off = 0;
  unsigned short* zbf = (unsigned short*)(ws + off); off += (size_t)M * L * 2;   // z (bf16)
  unsigned short* hbf = (unsigned short*)(ws + off); off += (size_t)M * H * 2;   // hidden (bf16)
  float*          gf  = (float*)(ws + off);          off += (size_t)M * H * 4;   // f32 GEMM out
  unsigned short* wbf = (unsigned short*)(ws + off); off += (size_t)H * 5000 * 2;// weights (bf16)
  float*          part= (float*)(ws + off);          off += (size_t)2 * NCH * 5000 * 4;
  float*          stats=(float*)(ws + off);          // 2*5000 floats

  const float* latent = (const float*)d_in[0];
  auto cdiv = [](int a, int b) { return (a + b - 1) / b; };

  // ---- GNN: fully-connected 4-node graph collapses all nodes to one feature ----
  int nz = M * L;
  k_sum_nodes<<<cdiv(nz, 256), 256, 0, stream>>>(latent, zbf, nz);

  for (int l = 0; l < 2; ++l) {
    const float* W = (const float*)d_in[1 + 2 * l];
    const float* b = (const float*)d_in[2 + 2 * l];
    k_cvt_bf16<<<cdiv(L * L, 256), 256, 0, stream>>>(W, wbf, L * L);
    k_gemm_bf16<<<dim3(cdiv(L, BN), M / BM), 256, 0, stream>>>(zbf, wbf, gf, M, L, L);
    k_bias_relu_bf16<<<cdiv(nz, 256), 256, 0, stream>>>(gf, b, zbf, nz, L);
  }

  // ---- generators; b1/b2 cancel inside training-mode BatchNorm, skipped ----
  float invM = 1.0f / (float)M;
  size_t outOff = 0;
  for (int t = 0; t < 4; ++t) {
    const float* W1  = (const float*)d_in[5 + 8 * t + 0];
    const float* g1  = (const float*)d_in[5 + 8 * t + 2];
    const float* be1 = (const float*)d_in[5 + 8 * t + 3];
    const float* W2  = (const float*)d_in[5 + 8 * t + 4];
    const float* g2  = (const float*)d_in[5 + 8 * t + 6];
    const float* be2 = (const float*)d_in[5 + 8 * t + 7];
    int D = dims[t];

    // layer 1: [M,L] @ [L,H] -> BN -> relu -> bf16
    k_cvt_bf16<<<cdiv(L * H, 256), 256, 0, stream>>>(W1, wbf, L * H);
    k_gemm_bf16<<<dim3(cdiv(H, BN), M / BM), 256, 0, stream>>>(zbf, wbf, gf, M, H, L);
    k_colstats_partial<<<dim3(cdiv(H, 256), NCH), 256, 0, stream>>>(gf, part, H, M / NCH, NCH);
    k_colstats_finalize<<<cdiv(H, 256), 256, 0, stream>>>(part, stats, H, NCH);
    k_bn_relu_bf16<<<cdiv(M * H, 256), 256, 0, stream>>>(gf, stats, g1, be1, hbf, M * H, H, invM);

    // layer 2: [M,H] @ [H,D] -> pre-BN straight into d_out slice, BN in place
    float* outp = (float*)d_out + outOff;
    k_cvt_bf16<<<cdiv(H * D, 256), 256, 0, stream>>>(W2, wbf, H * D);
    k_gemm_bf16<<<dim3(cdiv(D, BN), M / BM), 256, 0, stream>>>(hbf, wbf, outp, M, D, H);
    k_colstats_partial<<<dim3(cdiv(D, 256), NCH), 256, 0, stream>>>(outp, part, D, M / NCH, NCH);
    k_colstats_finalize<<<cdiv(D, 256), 256, 0, stream>>>(part, stats, D, NCH);
    k_bn_inplace<<<cdiv(M * D, 256), 256, 0, stream>>>(outp, stats, g2, be2, M * D, D, invM);

    outOff += (size_t)M * D;
  }
}
